// LSTFEncoder_3178275799753
// MI455X (gfx1250) — matrix operand order repaired and checked by reading, not compiled, not run on hardware
//
#include <hip/hip_runtime.h>
#include <cfloat>
#include <cmath>

typedef _Float16 f16;
typedef __attribute__((ext_vector_type(16))) _Float16 v16h;
typedef __attribute__((ext_vector_type(8)))  float    v8f;
typedef __attribute__((ext_vector_type(4)))  unsigned v4u;
typedef __attribute__((ext_vector_type(8)))  int      v8i;
typedef __attribute__((ext_vector_type(4)))  int      v4i;

#define NB  4
#define NH  8
#define DM  512
#define DH  64
#define DFF 2048

#if defined(__has_builtin)
#  if __has_builtin(__builtin_amdgcn_tensor_load_to_lds) && \
      __has_builtin(__builtin_amdgcn_s_wait_tensorcnt)
#    define USE_TDM 1
#  endif
#endif
#ifndef USE_TDM
#  define USE_TDM 0
#endif

static __device__ __forceinline__ unsigned wangHash(unsigned s) {
  s = (s ^ 61u) ^ (s >> 16); s *= 9u; s ^= s >> 4; s *= 0x27d4eb2du; s ^= s >> 15;
  return s;
}

#if USE_TDM
// Issue one TDM 2D tile load: 64 rows x 64 f16 cols from a row-major (rows, K)
// matrix into LDS at lds_addr. D# packing per CDNA5 ISA 8.3/8.4.
static __device__ __forceinline__ void tdm_load_tile(unsigned lds_addr,
                                                     const f16* gptr,
                                                     int K, int Nrows) {
  unsigned long long ga = (unsigned long long)(size_t)(const void*)gptr;
  v4u g0;
  g0[0] = 1u;                                    // count=1, user mode
  g0[1] = lds_addr;                              // lds_addr [63:32]
  g0[2] = (unsigned)(ga & 0xffffffffu);          // global_addr lo
  g0[3] = (unsigned)((ga >> 32) & 0x1ffffffu) | (2u << 30); // addr hi | type=2
  unsigned td0 = (unsigned)K;                    // tensor_dim0 (elements)
  unsigned td1 = (unsigned)Nrows;                // tensor_dim1
  unsigned long long st0 = (unsigned long long)K; // tensor_dim0_stride
  v8i g1;
  g1[0] = (int)(1u << 16);                       // workgroup_mask=0, data_size=1 (2B)
  g1[1] = (int)((td0 & 0xffffu) << 16);          // tensor_dim0 lo16 at bits 63:48
  g1[2] = (int)((td0 >> 16) | ((td1 & 0xffffu) << 16));
  g1[3] = (int)((td1 >> 16) | (64u << 16));      // tile_dim0=64
  g1[4] = (int)(64u);                            // tile_dim1=64, tile_dim2=0
  g1[5] = (int)(st0 & 0xffffffffu);
  g1[6] = (int)((st0 >> 32) & 0xffffu);          // stride1=0
  g1[7] = 0;
  v4i z4 = {0, 0, 0, 0};
#if __clang_major__ >= 23
  v8i z8 = {0, 0, 0, 0, 0, 0, 0, 0};
  __builtin_amdgcn_tensor_load_to_lds(g0, g1, z4, z4, z8, 0);
#else
  __builtin_amdgcn_tensor_load_to_lds(g0, g1, z4, z4, 0);
#endif
}
#endif

// ---------------- weight conversion ----------------
__global__ void k_convT(const float* __restrict__ src, f16* __restrict__ dst, int K, int N) {
  int gid = blockIdx.x * 256 + threadIdx.x;
  if (gid >= K * N) return;
  int n = gid / K, k = gid - n * K;
  dst[gid] = (f16)src[(size_t)k * N + n];
}

__global__ void k_convC(const float* __restrict__ src, f16* __restrict__ dst, int n) {
  int gid = blockIdx.x * 256 + threadIdx.x;
  if (gid >= n) return;
  dst[gid] = (f16)src[gid];
}

__global__ void k_convTap(const float* __restrict__ src, f16* __restrict__ dst) {
  int gid = blockIdx.x * 256 + threadIdx.x;
  if (gid >= DM * DM) return;
  #pragma unroll
  for (int t = 0; t < 3; ++t)
    dst[(size_t)t * DM * DM + gid] = (f16)src[(size_t)gid * 3 + t];
}

// ---------------- token embedding + positional encoding ----------------
__global__ void k_tok(const float* __restrict__ xe, const float* __restrict__ tw,
                      float* __restrict__ x, f16* __restrict__ xh, int L) {
  int gid = blockIdx.x * 256 + threadIdx.x;
  if (gid >= NB * L * DM) return;
  int o = gid % DM; int l = (gid / DM) % L; int b = gid / (DM * L);
  float acc = 0.f;
  #pragma unroll
  for (int t = 0; t < 3; ++t) {
    int lm = l + t - 1; if (lm < 0) lm += L; if (lm >= L) lm -= L;
    const float* xr = xe + ((size_t)b * L + lm) * 16;
    #pragma unroll
    for (int i = 0; i < 16; ++i) acc += tw[(o * 16 + i) * 3 + t] * xr[i];
  }
  int j = o >> 1;
  float div = expf(-(float)(2 * j) * (9.210340371976184f / 512.f));
  float ang = (float)l * div;
  acc += (o & 1) ? cosf(ang) : sinf(ang);
  x[gid] = acc;
  xh[gid] = (f16)acc;
}

// ---------------- WMMA GEMM ----------------
// D[m,n] = act( sum_k A[remap(m),k]*Bw[n,k] + bias[n] + Cin[m,n] )
// A f16 (M,K) with circular per-batch row shift; Bw f16 (N,K).
// Block: 256 threads (8 wave32), tile 64x64, K step 64.
// Each wave: one 16-row strip x two 16-col tiles -> 4 WMMA per K step.
__global__ __launch_bounds__(256) void k_gemm(
    const f16* __restrict__ A, const f16* __restrict__ Bw,
    const float* __restrict__ bias, const float* __restrict__ Cin,
    float* __restrict__ Cout, f16* __restrict__ Cout16,
    int M, int N, int K, int Lb, int shift, int act) {
  (void)M;
  __shared__ __align__(16) f16 As[64 * 64];
  __shared__ __align__(16) f16 Bs[64 * 64];
  const int t = threadIdx.x;
  const int lane = t & 31, wid = t >> 5;
  const int wm = wid >> 1, wn = wid & 1;
  const int lo = lane & 15, hi = lane >> 4;
  const int m0 = blockIdx.y * 64, n0 = blockIdx.x * 64;
  // cooperative load coordinates: 4 threads per 64-half row
  const int ar = t >> 2;
  const int ac = (t & 3) * 16;
  int gm = m0 + ar;
  int bb = gm / Lb, l = gm - bb * Lb;
  int ls = l + shift; if (ls < 0) ls += Lb; if (ls >= Lb) ls -= Lb;
  const f16* arow = A + ((size_t)bb * Lb + ls) * K + ac;
#if !USE_TDM
  const f16* brow = Bw + (size_t)(n0 + ar) * K + ac;
#endif
  v8f acc0 = {}, acc1 = {};
  for (int k0 = 0; k0 < K; k0 += 64) {
    const uint4* pa = (const uint4*)(arow + k0);
    uint4 a0 = pa[0], a1 = pa[1];
    *(uint4*)&As[ar * 64 + ac]     = a0;
    *(uint4*)&As[ar * 64 + ac + 8] = a1;
#if USE_TDM
    if (wid == 0)
      tdm_load_tile((unsigned)(size_t)(void*)&Bs[0], Bw + (size_t)n0 * K + k0, K, N);
    __builtin_amdgcn_s_wait_tensorcnt(0);
#else
    const uint4* pb = (const uint4*)(brow + k0);
    uint4 b0 = pb[0], b1 = pb[1];
    *(uint4*)&Bs[ar * 64 + ac]     = b0;
    *(uint4*)&Bs[ar * 64 + ac + 8] = b1;
#endif
    if (k0 + 64 < K) __builtin_prefetch(arow + k0 + 64, 0, 3);
    __syncthreads();
    #pragma unroll
    for (int s = 0; s < 2; ++s) {
      v16h fa, fb0, fb1;
      #pragma unroll
      for (int v = 0; v < 8; ++v) {
        // A (16x32 f16): lane lo = M; K = base + (v<4?0:16) + hi*8 + (v%4)*2 + h
        int ka = s * 32 + ((v < 4) ? 0 : 16) + hi * 8 + (v & 3) * 2;
        int ia = (wm * 16 + lo) * 64 + ka;
        fa[2 * v]     = As[ia];
        fa[2 * v + 1] = As[ia + 1];
        // B (32x16 f16): lane lo = N; K = base + (v<4?0:16) + hi*8 + (v%4)*2 + h
        int kb = s * 32 + ((v < 4) ? 0 : 16) + hi * 8 + (v & 3) * 2;
        int ib0 = (wn * 32 + lo) * 64 + kb;
        int ib1 = (wn * 32 + 16 + lo) * 64 + kb;
        fb0[2 * v]     = Bs[ib0];
        fb0[2 * v + 1] = Bs[ib0 + 1];
        fb1[2 * v]     = Bs[ib1];
        fb1[2 * v + 1] = Bs[ib1 + 1];
      }
      acc0 = __builtin_amdgcn_wmma_f32_16x16x32_f16(false, fa, false, fb0, (short)0,
                                                    acc0, false, false);
      acc1 = __builtin_amdgcn_wmma_f32_16x16x32_f16(false, fa, false, fb1, (short)0,
                                                    acc1, false, false);
    }
    __syncthreads();
  }
  // C/D layout: lane lo = N col, VGPR r -> M = hi*8 + r
  #pragma unroll
  for (int nt = 0; nt < 2; ++nt) {
    v8f acc = nt ? acc1 : acc0;
    int n = n0 + wn * 32 + nt * 16 + lo;
    float bval = bias ? bias[n] : 0.f;
    #pragma unroll
    for (int r = 0; r < 8; ++r) {
      int m = m0 + wm * 16 + hi * 8 + r;
      float val = acc[r] + bval;
      if (Cin) val += Cin[(size_t)m * N + n];
      if (act == 1) val = 0.5f * val * (1.f + erff(val * 0.7071067811865475f));
      if (Cout)   Cout[(size_t)m * N + n]   = val;
      if (Cout16) Cout16[(size_t)m * N + n] = (f16)val;
    }
  }
}

// ---------------- ProbSparse attention pieces ----------------
__global__ void k_probM(const float* __restrict__ q, const float* __restrict__ kk,
                        float* __restrict__ M, int Lq, int Lk, int U, int layer) {
  int gid = blockIdx.x * 256 + threadIdx.x;
  if (gid >= NB * NH * Lq) return;
  int l = gid % Lq; int h = (gid / Lq) % NH; int b = gid / (Lq * NH);
  const float* qr = q + (((size_t)b * Lq + l) * NH + h) * DH;
  float mx = -FLT_MAX, sm = 0.f;
  for (int u = 0; u < U; ++u) {
    unsigned hsh = wangHash((unsigned)l * 2654435761u ^ (unsigned)u * 40503u ^
                            (unsigned)layer * 0x9E3779B9u);
    int ki = (int)(hsh % (unsigned)Lk);
    const float* kr = kk + (((size_t)b * Lk + ki) * NH + h) * DH;
    float d = 0.f;
    #pragma unroll 8
    for (int e = 0; e < DH; ++e) d += qr[e] * kr[e];
    mx = fmaxf(mx, d); sm += d;
  }
  M[gid] = mx - sm / (float)Lk;
}

__global__ void k_topk(float* __restrict__ M, int* __restrict__ top, int Lq, int U) {
  __shared__ float sv[256];
  __shared__ int   si[256];
  int bh = blockIdx.x; int t = threadIdx.x;
  float* Mp = M + (size_t)bh * Lq;
  for (int it = 0; it < U; ++it) {
    float best = -FLT_MAX; int bi = 0x7fffffff;
    for (int l = t; l < Lq; l += 256) {
      float v = Mp[l];
      if (v > best || (v == best && l < bi)) { best = v; bi = l; }
    }
    sv[t] = best; si[t] = bi; __syncthreads();
    for (int s = 128; s > 0; s >>= 1) {
      if (t < s) {
        float v2 = sv[t + s]; int i2 = si[t + s];
        if (v2 > sv[t] || (v2 == sv[t] && i2 < si[t])) { sv[t] = v2; si[t] = i2; }
      }
      __syncthreads();
    }
    if (t == 0) { top[bh * U + it] = si[0]; Mp[si[0]] = -FLT_MAX; }
    __syncthreads();
  }
}

__global__ void k_scores(const float* __restrict__ q, const float* __restrict__ kk,
                         const int* __restrict__ top, float* __restrict__ sc,
                         int Lq, int Lk, int U) {
  int gid = blockIdx.x * 256 + threadIdx.x;
  if (gid >= NB * NH * U * Lk) return;
  int l = gid % Lk; int uu = (gid / Lk) % U; int bh = gid / (Lk * U);
  int b = bh / NH, h = bh % NH;
  int ql = top[bh * U + uu];
  const float* qr = q + (((size_t)b * Lq + ql) * NH + h) * DH;
  const float* kr = kk + (((size_t)b * Lk + l) * NH + h) * DH;
  float d = 0.f;
  #pragma unroll 8
  for (int e = 0; e < DH; ++e) d += qr[e] * kr[e];
  sc[gid] = d * 0.125f; // 1/sqrt(64)
}

__global__ void k_softmax(float* __restrict__ sc, int Lk) {
  int row = blockIdx.x, t = threadIdx.x;
  float* p = sc + (size_t)row * Lk;
  __shared__ float red[256];
  float mx = -FLT_MAX;
  for (int l = t; l < Lk; l += 256) mx = fmaxf(mx, p[l]);
  red[t] = mx; __syncthreads();
  for (int s = 128; s > 0; s >>= 1) { if (t < s) red[t] = fmaxf(red[t], red[t + s]); __syncthreads(); }
  mx = red[0]; __syncthreads();
  float sm = 0.f;
  for (int l = t; l < Lk; l += 256) { float e = expf(p[l] - mx); p[l] = e; sm += e; }
  red[t] = sm; __syncthreads();
  for (int s = 128; s > 0; s >>= 1) { if (t < s) red[t] += red[t + s]; __syncthreads(); }
  float inv = 1.f / red[0];
  for (int l = t; l < Lk; l += 256) p[l] *= inv;
}

__global__ void k_vmean(const float* __restrict__ v, float* __restrict__ vm, int Lk) {
  int gid = blockIdx.x * 256 + threadIdx.x;
  if (gid >= NB * NH * DH) return;
  int d = gid % DH; int bh = gid / DH; int b = bh / NH, h = bh % NH;
  float s = 0.f;
  for (int l = 0; l < Lk; ++l) s += v[(((size_t)b * Lk + l) * NH + h) * DH + d];
  vm[gid] = s / (float)Lk;
}

__global__ void k_fill(const float* __restrict__ vm, f16* __restrict__ attn, int Lq) {
  int gid = blockIdx.x * 256 + threadIdx.x;
  if (gid >= NB * Lq * DM) return;
  int cc = gid % DM; int b = gid / (DM * Lq);
  int h = cc / DH, d = cc % DH;
  attn[gid] = (f16)vm[(b * NH + h) * DH + d];
}

__global__ void k_upd(const float* __restrict__ sc, const float* __restrict__ v,
                      const int* __restrict__ top, f16* __restrict__ attn,
                      int Lq, int Lk, int U) {
  int gid = blockIdx.x * 256 + threadIdx.x;
  if (gid >= NB * NH * U * DH) return;
  int d = gid % DH; int uu = (gid / DH) % U; int bh = gid / (DH * U);
  int b = bh / NH, h = bh % NH;
  const float* srow = sc + ((size_t)bh * U + uu) * Lk;
  float acc = 0.f;
  for (int l = 0; l < Lk; ++l)
    acc += srow[l] * v[(((size_t)b * Lk + l) * NH + h) * DH + d];
  int ql = top[bh * U + uu];
  attn[((size_t)b * Lq + ql) * DM + h * DH + d] = (f16)acc;
}

// ---------------- normalization / distill ----------------
__global__ void k_lnorm(const float* __restrict__ in, const float* __restrict__ g,
                        const float* __restrict__ b, float* __restrict__ out,
                        f16* __restrict__ out16) {
  int row = blockIdx.x, t = threadIdx.x;
  const float* p = in + (size_t)row * DM;
  __shared__ float red[256];
  float a0 = p[t], a1 = p[t + 256];
  red[t] = a0 + a1; __syncthreads();
  for (int s = 128; s > 0; s >>= 1) { if (t < s) red[t] += red[t + s]; __syncthreads(); }
  float mean = red[0] * (1.f / DM); __syncthreads();
  float d0 = a0 - mean, d1 = a1 - mean;
  red[t] = d0 * d0 + d1 * d1; __syncthreads();
  for (int s = 128; s > 0; s >>= 1) { if (t < s) red[t] += red[t + s]; __syncthreads(); }
  float inv = rsqrtf(red[0] * (1.f / DM) + 1e-5f);
  float o0 = d0 * inv * g[t] + b[t];
  float o1 = d1 * inv * g[t + 256] + b[t + 256];
  out[(size_t)row * DM + t]       = o0;
  out[(size_t)row * DM + t + 256] = o1;
  if (out16) {
    out16[(size_t)row * DM + t]       = (f16)o0;
    out16[(size_t)row * DM + t + 256] = (f16)o1;
  }
}

__global__ void k_bnstats(const float* __restrict__ y, float* __restrict__ mean,
                          float* __restrict__ var, int Mr) {
  int c = blockIdx.x, t = threadIdx.x;
  __shared__ float rs[256], rss[256];
  float s = 0.f, ss = 0.f;
  for (int r = t; r < Mr; r += 256) { float v = y[(size_t)r * DM + c]; s += v; ss += v * v; }
  rs[t] = s; rss[t] = ss; __syncthreads();
  for (int st = 128; st > 0; st >>= 1) {
    if (t < st) { rs[t] += rs[t + st]; rss[t] += rss[t + st]; }
    __syncthreads();
  }
  if (t == 0) { float m = rs[0] / Mr; mean[c] = m; var[c] = rss[0] / Mr - m * m; }
}

__global__ void k_bnelu(float* __restrict__ y, const float* __restrict__ mean,
                        const float* __restrict__ var, const float* __restrict__ g,
                        const float* __restrict__ b, int n) {
  int gid = blockIdx.x * 256 + threadIdx.x;
  if (gid >= n) return;
  int c = gid % DM;
  float val = (y[gid] - mean[c]) * rsqrtf(var[c] + 1e-5f) * g[c] + b[c];
  y[gid] = val > 0.f ? val : expm1f(val);
}

__global__ void k_maxpool(const float* __restrict__ y, float* __restrict__ out,
                          f16* __restrict__ out16, int Lc) {
  int Lo = Lc >> 1;
  int gid = blockIdx.x * 256 + threadIdx.x;
  if (gid >= NB * Lo * DM) return;
  int c = gid % DM; int lo = (gid / DM) % Lo; int b = gid / (DM * Lo);
  size_t base = (size_t)b * Lc * DM + c;
  int l = 2 * lo;
  float m = y[base + (size_t)l * DM];
  if (l >= 1)     m = fmaxf(m, y[base + (size_t)(l - 1) * DM]);
  if (l + 1 < Lc) m = fmaxf(m, y[base + (size_t)(l + 1) * DM]);
  out[gid] = m;
  out16[gid] = (f16)m;
}

// ---------------- host orchestration ----------------
extern "C" void kernel_launch(void* const* d_in, const int* in_sizes, int n_in,
                              void* d_out, int out_size, void* d_ws, size_t ws_size,
                              hipStream_t stream) {
  (void)in_sizes; (void)n_in; (void)out_size; (void)ws_size;
  const float* xe    = (const float*)d_in[0];
  const float* tokw  = (const float*)d_in[1];
  const float* normg = (const float*)d_in[58];
  const float* normb = (const float*)d_in[59];
#define LP(i, j) ((const float*)d_in[2 + 16 * (i) + (j)])
#define CP(j, k) ((const float*)d_in[50 + 4 * (j) + (k)])

  char* wsp = (char*)d_ws;
  size_t off = 0;
  auto alloc = [&](size_t bytes) -> void* {
    void* p = wsp + off; off += (bytes + 255) & ~(size_t)255; return p;
  };
  const size_t AMAX = (size_t)NB * 2048 * DM;
  float* xbuf = (float*)alloc(AMAX * 4);
  float* ybuf = (float*)alloc(AMAX * 4);
  float* qb   = (float*)alloc(AMAX * 4);
  float* kb   = (float*)alloc(AMAX * 4);
  float* vb   = (float*)alloc(AMAX * 4);
  f16*   xh   = (f16*)alloc(AMAX * 2);
  f16*   a16  = (f16*)alloc(AMAX * 2);
  f16*   h116 = (f16*)alloc((size_t)NB * 2048 * DFF * 2);
  float* sc   = (float*)alloc((size_t)NB * NH * 40 * 2048 * 4);
  float* Mb   = (float*)alloc((size_t)NB * NH * 2048 * 4);
  int*   tk   = (int*)alloc((size_t)NB * NH * 64 * 4);
  float* vm   = (float*)alloc((size_t)NB * NH * DH * 4);
  float* bnm  = (float*)alloc(DM * 4);
  float* bnv  = (float*)alloc(DM * 4);
  const size_t LWsz = (size_t)4 * DM * DM + (size_t)2 * DM * DFF; // f16 elems / layer
  f16* wh = (f16*)alloc((3 * LWsz + (size_t)2 * 3 * DM * DM) * 2);

  const int thr = 256;
  for (int i = 0; i < 3; ++i) {
    f16* base = wh + (size_t)i * LWsz;
    k_convT<<<(DM * DM + 255) / 256, thr, 0, stream>>>(LP(i, 0), base + 0 * DM * DM, DM, DM);
    k_convT<<<(DM * DM + 255) / 256, thr, 0, stream>>>(LP(i, 2), base + 1 * DM * DM, DM, DM);
    k_convT<<<(DM * DM + 255) / 256, thr, 0, stream>>>(LP(i, 4), base + 2 * DM * DM, DM, DM);
    k_convT<<<(DM * DM + 255) / 256, thr, 0, stream>>>(LP(i, 6), base + 3 * DM * DM, DM, DM);
    k_convC<<<(DFF * DM + 255) / 256, thr, 0, stream>>>(LP(i, 8), base + 4 * DM * DM, DFF * DM);
    k_convC<<<(DM * DFF + 255) / 256, thr, 0, stream>>>(LP(i, 10), base + 4 * DM * DM + DFF * DM, DM * DFF);
  }
  f16* whc = wh + 3 * LWsz;
  for (int j = 0; j < 2; ++j)
    k_convTap<<<(DM * DM + 255) / 256, thr, 0, stream>>>(CP(j, 0), whc + (size_t)j * 3 * DM * DM);

  k_tok<<<(NB * 2048 * DM + 255) / 256, thr, 0, stream>>>(xe, tokw, xbuf, xh, 2048);

  auto gemm = [&](const f16* A, const f16* Bw, const float* bias, const float* Cin,
                  float* Cout, f16* Cout16, int Mr, int Nn, int Kk, int Lb,
                  int shift, int act) {
    dim3 grid(Nn / 64, Mr / 64);
    k_gemm<<<grid, 256, 0, stream>>>(A, Bw, bias, Cin, Cout, Cout16, Mr, Nn, Kk, Lb,
                                     shift, act);
  };

  int Lc = 2048;
  for (int i = 0; i < 3; ++i) {
    int Mr = NB * Lc;
    int U = 5 * (int)ceil(log((double)Lc)); if (U > Lc) U = Lc;
    f16* base = wh + (size_t)i * LWsz;
    const f16 *wq = base, *wk = base + DM * DM, *wv = base + 2 * DM * DM,
              *wo = base + 3 * DM * DM, *w1 = base + 4 * DM * DM,
              *w2 = base + 4 * DM * DM + DFF * DM;

    gemm(xh, wq, LP(i, 1), nullptr, qb, nullptr, Mr, DM, DM, Lc, 0, 0);
    gemm(xh, wk, LP(i, 3), nullptr, kb, nullptr, Mr, DM, DM, Lc, 0, 0);
    gemm(xh, wv, LP(i, 5), nullptr, vb, nullptr, Mr, DM, DM, Lc, 0, 0);

    k_probM<<<(NB * NH * Lc + 255) / 256, thr, 0, stream>>>(qb, kb, Mb, Lc, Lc, U, i);
    k_topk<<<NB * NH, thr, 0, stream>>>(Mb, tk, Lc, U);
    k_scores<<<(NB * NH * U * Lc + 255) / 256, thr, 0, stream>>>(qb, kb, tk, sc, Lc, Lc, U);
    k_softmax<<<NB * NH * U, thr, 0, stream>>>(sc, Lc);
    k_vmean<<<(NB * NH * DH + 255) / 256, thr, 0, stream>>>(vb, vm, Lc);
    k_fill<<<(Mr * DM + 255) / 256, thr, 0, stream>>>(vm, a16, Lc);
    k_upd<<<(NB * NH * U * DH + 255) / 256, thr, 0, stream>>>(sc, vb, tk, a16, Lc, Lc, U);

    gemm(a16, wo, LP(i, 7), xbuf, ybuf, nullptr, Mr, DM, DM, Lc, 0, 0);
    k_lnorm<<<Mr, thr, 0, stream>>>(ybuf, LP(i, 12), LP(i, 13), xbuf, xh);

    gemm(xh, w1, LP(i, 9), nullptr, nullptr, h116, Mr, DFF, DM, Lc, 0, 1); // GELU
    gemm(h116, w2, LP(i, 11), xbuf, ybuf, nullptr, Mr, DM, DFF, Lc, 0, 0); // +resid
    k_lnorm<<<Mr, thr, 0, stream>>>(ybuf, LP(i, 14), LP(i, 15), xbuf, xh);

    if (i < 2) {
      f16* ct = whc + (size_t)i * 3 * DM * DM;
      gemm(xh, ct + 0 * DM * DM, CP(i, 1), nullptr, ybuf, nullptr, Mr, DM, DM, Lc, -1, 0);
      gemm(xh, ct + 1 * DM * DM, nullptr, ybuf, ybuf, nullptr, Mr, DM, DM, Lc, 0, 0);
      gemm(xh, ct + 2 * DM * DM, nullptr, ybuf, ybuf, nullptr, Mr, DM, DM, Lc, +1, 0);
      k_bnstats<<<DM, thr, 0, stream>>>(ybuf, bnm, bnv, Mr);
      k_bnelu<<<(Mr * DM + 255) / 256, thr, 0, stream>>>(ybuf, bnm, bnv, CP(i, 2), CP(i, 3), Mr * DM);
      k_maxpool<<<((Mr / 2) * DM + 255) / 256, thr, 0, stream>>>(ybuf, xbuf, xh, Lc);
      Lc >>= 1;
    }
  }
  k_lnorm<<<NB * Lc, thr, 0, stream>>>(xbuf, normg, normb, (float*)d_out, nullptr);
#undef LP
#undef CP
}
